// TemporalAttention_61924838473817
// MI455X (gfx1250) — matrix operand order repaired and checked by reading, not compiled, not run on hardware
//
#include <hip/hip_runtime.h>
#include <hip/hip_bf16.h>

// ---------------------------------------------------------------------------
// Temporal attention for x:[B=8,S=128,N=500,F=64], W*:[64,64], b*:[64]
// out:[B,N,F] = sum_s softmax_t(Q K^T / sqrt(F))[s,t] @ V
// One workgroup (256 threads = 8 wave32) per (b,n) pair; all GEMMs via
// v_wmma_f32_16x16x32_bf16 with fp32 accumulation. float4 global fetch,
// packed bf16 LDS staging, deferred softmax normalization, ds_add_f32
// final reduction.
// ---------------------------------------------------------------------------

typedef __attribute__((ext_vector_type(16))) __bf16 v16bf;
typedef __attribute__((ext_vector_type(8)))  float  v8f;

#define Bdim 8
#define Sdim 128
#define Ndim 500
#define Fdim 64

// LDS strides in elements; all row pitches are multiples of 8 bytes so that
// b64 stores / 2addr_b64 loads stay aligned.
#define XLD  68   // x slice  [128][64] bf16 (pitch 136B)
#define WLD  68   // weights  [64][64]  bf16 (pitch 136B), x3
#define QLD  68   // Q        [128][64] bf16
#define KLD  68   // K        [128][64] bf16
#define VTLD 132  // V^T      [64][128] bf16 (pitch 264B)
#define SLD  132  // scores/P [128][128] bf16 (pitch 264B), aliases X region

// byte offsets into one shared buffer
#define OFF_A     0                       // region A: X (17408B) then scores (33792B)
#define SZ_A      (128 * SLD * 2)         // 33792
#define OFF_W     (OFF_A + SZ_A)
#define SZ_W      (3 * 64 * WLD * 2)      // 26112
#define OFF_BIAS  (OFF_W + SZ_W)
#define SZ_BIAS   (3 * 64 * 4)            // 768
#define OFF_Q     (OFF_BIAS + SZ_BIAS)
#define SZ_Q      (128 * QLD * 2)         // 17408
#define OFF_K     (OFF_Q + SZ_Q)
#define SZ_K      (128 * KLD * 2)         // 17408
#define OFF_VT    (OFF_K + SZ_K)
#define SZ_VT     (64 * VTLD * 2)         // 16896
#define OFF_IZ    (OFF_VT + SZ_VT)
#define SZ_IZ     (128 * 4)               // 512
#define OFF_ACC   (OFF_IZ + SZ_IZ)
#define SZ_ACC    (64 * 4)                // 256
#define SMEM_BYTES (OFF_ACC + SZ_ACC)     // 113152 bytes (< 320KB WGP LDS)

__device__ __forceinline__ __bf16 f2bf(float f) { return (__bf16)f; }
__device__ __forceinline__ float  bf2f(__bf16 b) { return (float)b; }

__device__ __forceinline__ uint2 pack4bf(float a, float b, float c, float d) {
  union { __bf16 h[4]; uint2 u; } p;
  p.h[0] = f2bf(a); p.h[1] = f2bf(b); p.h[2] = f2bf(c); p.h[3] = f2bf(d);
  return p.u;
}

// A fragment (16x32 bf16) from a row-major bf16 LDS matrix.
// Layout: lane l<16 -> row M=l, K={k0+0..7, k0+16..23}; lane l+16 -> K={+8..15, +24..31}
__device__ __forceinline__ v16bf frag_A(const __bf16* base, int ld, int row0, int k0, int lane) {
  const int half = lane >> 4;
  const __bf16* rowp = base + (row0 + (lane & 15)) * ld;
  union { v16bf v; unsigned int u[8]; } f;
#pragma unroll
  for (int j = 0; j < 8; ++j) {
    const int k = k0 + ((j < 4) ? 0 : 16) + half * 8 + 2 * (j & 3);
    f.u[j] = *(const unsigned int*)(rowp + k);
  }
  return f.v;
}

// B fragment (32x16 bf16) where B[k][n] = Mat[n][k] (Mat row-major in LDS).
// Layout: lane l<16 -> col N=l, K={k0+0..7, k0+16..23}; lane l+16 -> col N=l-16, K={+8..15, +24..31}
__device__ __forceinline__ v16bf frag_BT(const __bf16* base, int ld, int n0, int k0, int lane) {
  const int half = lane >> 4;
  const __bf16* rowp = base + (n0 + (lane & 15)) * ld;
  union { v16bf v; unsigned int u[8]; } f;
#pragma unroll
  for (int j = 0; j < 8; ++j) {
    const int k = k0 + ((j < 4) ? 0 : 16) + half * 8 + 2 * (j & 3);
    f.u[j] = *(const unsigned int*)(rowp + k);
  }
  return f.v;
}

__device__ __forceinline__ v8f wmma_bf16(v16bf a, v16bf b, v8f c) {
  return __builtin_amdgcn_wmma_f32_16x16x32_bf16(false, a, false, b,
                                                 (short)0, c, false, false);
}

__global__ __launch_bounds__(256, 1)
void temporal_attn_wmma_bf16(const float* __restrict__ x,
                             const float* __restrict__ Wq, const float* __restrict__ bq,
                             const float* __restrict__ Wk, const float* __restrict__ bk,
                             const float* __restrict__ Wv, const float* __restrict__ bv,
                             float* __restrict__ out) {
  __shared__ __align__(16) unsigned char smem[SMEM_BYTES];
  __bf16* Xl   = (__bf16*)(smem + OFF_A);    // stages 1-2
  __bf16* Pl   = (__bf16*)(smem + OFF_A);    // stages 3-5 (aliases X)
  __bf16* Wl   = (__bf16*)(smem + OFF_W);    // 3 matrices, each 64*WLD
  float*  bl   = (float*)(smem + OFF_BIAS);  // 3 x 64
  __bf16* Ql   = (__bf16*)(smem + OFF_Q);
  __bf16* Kl   = (__bf16*)(smem + OFF_K);
  __bf16* Vtl  = (__bf16*)(smem + OFF_VT);   // [f][t]
  float*  invZ = (float*)(smem + OFF_IZ);
  float*  oacc = (float*)(smem + OFF_ACC);

  const int tid  = threadIdx.x;
  const int wv   = tid >> 5;
  const int lane = tid & 31;
  const int bid  = blockIdx.x;        // b*Ndim + n
  const int bb   = bid / Ndim;
  const int nn   = bid - bb * Ndim;

  if (tid < 64) oacc[tid] = 0.0f;

  // -------- Stage 1: stage x slice + weights + biases into LDS as bf16 -----
  // x slice: 128 rows x 16 float4 -> global_load_b128 + packed b64 LDS stores
  for (int idx = tid; idx < Sdim * (Fdim / 4); idx += 256) {
    const int s = idx >> 4, f = (idx & 15) * 4;
    const size_t gi = (((size_t)bb * Sdim + s) * Ndim + nn) * Fdim + f;
    const float4 v = *(const float4*)(x + gi);
    *(uint2*)(Xl + s * XLD + f) = pack4bf(v.x, v.y, v.z, v.w);
  }
  // weights: 3 * 64 rows x 16 float4
  for (int idx = tid; idx < 3 * 64 * (Fdim / 4); idx += 256) {
    const int p = idx >> 10, rem = idx & 1023, r = rem >> 4, c = (rem & 15) * 4;
    const float* Wsrc = (p == 0) ? Wq : ((p == 1) ? Wk : Wv);
    const float4 v = *(const float4*)(Wsrc + r * 64 + c);
    *(uint2*)(Wl + p * 64 * WLD + r * WLD + c) = pack4bf(v.x, v.y, v.z, v.w);
  }
  if (tid < 192) {
    const int p = tid >> 6, c = tid & 63;
    const float* bsrc = (p == 0) ? bq : ((p == 1) ? bk : bv);
    bl[tid] = bsrc[c];
  }
  __syncthreads();

  // -------- Stage 2: Q/K/V projections: [128x64] = X @ W^T + b -------------
  // 8 Mtiles * 4 Ftiles = 32 tile jobs per projection; three branch-free loops.
  // Q
  for (int job = wv; job < 32; job += 8) {
    const int mt = job >> 2, ft = job & 3;
    v8f acc = {};
#pragma unroll
    for (int kt = 0; kt < 2; ++kt)
      acc = wmma_bf16(frag_A(Xl, XLD, mt * 16, kt * 32, lane),
                      frag_BT(Wl, WLD, ft * 16, kt * 32, lane), acc);
    const int f = ft * 16 + (lane & 15);
    const float bias = bl[f];
    const int mbase = mt * 16 + (lane >> 4) * 8;
#pragma unroll
    for (int r = 0; r < 8; ++r) Ql[(mbase + r) * QLD + f] = f2bf(acc[r] + bias);
  }
  // K
  for (int job = wv; job < 32; job += 8) {
    const int mt = job >> 2, ft = job & 3;
    v8f acc = {};
#pragma unroll
    for (int kt = 0; kt < 2; ++kt)
      acc = wmma_bf16(frag_A(Xl, XLD, mt * 16, kt * 32, lane),
                      frag_BT(Wl + 64 * WLD, WLD, ft * 16, kt * 32, lane), acc);
    const int f = ft * 16 + (lane & 15);
    const float bias = bl[64 + f];
    const int mbase = mt * 16 + (lane >> 4) * 8;
#pragma unroll
    for (int r = 0; r < 8; ++r) Kl[(mbase + r) * KLD + f] = f2bf(acc[r] + bias);
  }
  // V (stored transposed: Vt[f][t])
  for (int job = wv; job < 32; job += 8) {
    const int mt = job >> 2, ft = job & 3;
    v8f acc = {};
#pragma unroll
    for (int kt = 0; kt < 2; ++kt)
      acc = wmma_bf16(frag_A(Xl, XLD, mt * 16, kt * 32, lane),
                      frag_BT(Wl + 2 * 64 * WLD, WLD, ft * 16, kt * 32, lane), acc);
    const int f = ft * 16 + (lane & 15);
    const float bias = bl[128 + f];
    const int mbase = mt * 16 + (lane >> 4) * 8;
#pragma unroll
    for (int r = 0; r < 8; ++r) Vtl[f * VTLD + (mbase + r)] = f2bf(acc[r] + bias);
  }
  __syncthreads();

  // -------- Stage 3: scores = (Q @ K^T) * 1/sqrt(64), stored bf16 ----------
  for (int job = wv; job < 64; job += 8) {
    const int mt = job >> 3, tt = job & 7;
    v8f acc = {};
#pragma unroll
    for (int kt = 0; kt < 2; ++kt)
      acc = wmma_bf16(frag_A(Ql, QLD, mt * 16, kt * 32, lane),
                      frag_BT(Kl, KLD, tt * 16, kt * 32, lane), acc);
    const int t = tt * 16 + (lane & 15);
    const int mbase = mt * 16 + (lane >> 4) * 8;
#pragma unroll
    for (int r = 0; r < 8; ++r) Pl[(mbase + r) * SLD + t] = f2bf(acc[r] * 0.125f);
  }
  __syncthreads();

  // -------- Stage 4: row-wise unnormalized softmax, keep 1/Z per row -------
  if (tid < Sdim) {
    __bf16* row = Pl + tid * SLD;
    float m = -3.0e38f;
    for (int t = 0; t < Sdim; ++t) m = fmaxf(m, bf2f(row[t]));
    float Z = 0.0f;
    for (int t = 0; t < Sdim; ++t) {
      const float e = __expf(bf2f(row[t]) - m);
      Z += e;
      row[t] = f2bf(e);
    }
    invZ[tid] = 1.0f / Z;
  }
  __syncthreads();

  // -------- Stage 5: O = P @ V, fold 1/Z and sum over s into oacc ----------
  for (int job = wv; job < 32; job += 8) {
    const int mt = job >> 2, ft = job & 3;
    v8f acc = {};
#pragma unroll
    for (int kt = 0; kt < 4; ++kt)
      acc = wmma_bf16(frag_A(Pl, SLD, mt * 16, kt * 32, lane),
                      frag_BT(Vtl, VTLD, ft * 16, kt * 32, lane), acc);  // B[t][f] = Vt[f][t]
    const int f = ft * 16 + (lane & 15);
    const int mbase = mt * 16 + (lane >> 4) * 8;
    float partial = 0.0f;
#pragma unroll
    for (int r = 0; r < 8; ++r) partial += acc[r] * invZ[mbase + r];
    atomicAdd(&oacc[f], partial);   // ds_add_f32
  }
  __syncthreads();

  // -------- Stage 6: write out[b,n,:] --------------------------------------
  if (tid < 64) out[(size_t)bid * Fdim + tid] = oacc[tid];
}

extern "C" void kernel_launch(void* const* d_in, const int* in_sizes, int n_in,
                              void* d_out, int out_size, void* d_ws, size_t ws_size,
                              hipStream_t stream) {
  const float* x  = (const float*)d_in[0];
  const float* Wq = (const float*)d_in[1];
  const float* bq = (const float*)d_in[2];
  const float* Wk = (const float*)d_in[3];
  const float* bk = (const float*)d_in[4];
  const float* Wv = (const float*)d_in[5];
  const float* bv = (const float*)d_in[6];
  float* out = (float*)d_out;

  dim3 grid(Bdim * Ndim);   // 4000 workgroups, one per (b, n)
  dim3 block(256);          // 8 wave32
  temporal_attn_wmma_bf16<<<grid, block, 0, stream>>>(x, Wq, bq, Wk, bk, Wv, bv, out);
}
